// _TransformerBlock_88364657147962
// MI455X (gfx1250) — matrix operand order repaired and checked by reading, not compiled, not run on hardware
//
#include <hip/hip_runtime.h>
#include <hip/hip_bf16.h>

#define DEV __device__ __forceinline__

typedef __attribute__((ext_vector_type(16))) __bf16 v16bf;
typedef __attribute__((ext_vector_type(8)))  float  v8f;

constexpr int Dm = 1024;   // model dim
constexpr int HA = 128;    // head dim (A)
constexpr int NH = 8;      // heads
constexpr int FF = 4096;
constexpr int Bb = 4;
constexpr int Ss = 2048;
constexpr int BS = Bb * Ss;

// ---------- bf16 helpers (round-to-nearest-even) ----------
DEV unsigned short f2bfbits(float f) {
  unsigned int u = __float_as_uint(f);
  u += 0x7FFFu + ((u >> 16) & 1u);
  return (unsigned short)(u >> 16);
}

struct Frag { unsigned int d[8]; };
DEV v16bf frag_cast(Frag f) { return __builtin_bit_cast(v16bf, f); }

// ---------- WMMA fragment loaders (CDNA5 wave32 layouts, packed dwords) ----
// A-matrix 16x32 bf16: lane holds row (lane&15); slot pair (VGPR v) starts at
// K = ((v<4)?0:16) + 8*half + 2*(v&3)  -> dword index (v<4?0:8)+4*half+(v&3)
DEV v16bf load_a_bf16(const unsigned short* Ap, int lda, int m0, int k0) {
  const int lane = threadIdx.x & 31, hlf = lane >> 4;
  const unsigned int* p =
      (const unsigned int*)(Ap + (size_t)(m0 + (lane & 15)) * lda + k0);
  Frag f;
#pragma unroll
  for (int v = 0; v < 4; ++v) {
    f.d[v]     = p[4 * hlf + v];
    f.d[4 + v] = p[8 + 4 * hlf + v];
  }
  return frag_cast(f);
}
// B-matrix KxN bf16 where B[k][n] = M[n0+n][k0+k] (same K order as A:
// slot i -> K = ((i<8)?0:16) + 8*half + (i&7)) : 8 dwords per lane
DEV v16bf load_bT_bf16(const unsigned short* Mp, int ldm, int n0, int k0) {
  const int lane = threadIdx.x & 31, hlf = lane >> 4;
  const unsigned int* p =
      (const unsigned int*)(Mp + (size_t)(n0 + (lane & 15)) * ldm + k0);
  Frag f;
#pragma unroll
  for (int j = 0; j < 4; ++j) {
    f.d[j]     = p[4 * hlf + j];
    f.d[4 + j] = p[8 + 4 * hlf + j];
  }
  return frag_cast(f);
}

DEV v8f wmma_bf16(v16bf a, v16bf b, v8f c) {
  return __builtin_amdgcn_wmma_f32_16x16x32_bf16(false, a, false, b, (short)0, c,
                                                 false, false);
}

// Software-pipelined 16x64 GEMM tile: ping-pong register double-buffering
// (k unrolled by 64 so buffers alternate roles with no register copies).
template <int KTOT>
DEV void gemm16x64(const unsigned short* __restrict__ Ap, int lda, int m0,
                   const unsigned short* __restrict__ Bp, int ldb, int n0,
                   v8f acc[4]) {
  static_assert(KTOT % 64 == 0, "K must be a multiple of 64");
  v16bf a0 = load_a_bf16(Ap, lda, m0, 0);
  v16bf b0[4];
#pragma unroll
  for (int t = 0; t < 4; ++t) b0[t] = load_bT_bf16(Bp, ldb, n0 + 16 * t, 0);
  v16bf a1;
  v16bf b1[4];
#pragma unroll 1
  for (int k0 = 0; k0 < KTOT - 64; k0 += 64) {
    a1 = load_a_bf16(Ap, lda, m0, k0 + 32);
#pragma unroll
    for (int t = 0; t < 4; ++t)
      b1[t] = load_bT_bf16(Bp, ldb, n0 + 16 * t, k0 + 32);
#pragma unroll
    for (int t = 0; t < 4; ++t) acc[t] = wmma_bf16(a0, b0[t], acc[t]);
    a0 = load_a_bf16(Ap, lda, m0, k0 + 64);
#pragma unroll
    for (int t = 0; t < 4; ++t)
      b0[t] = load_bT_bf16(Bp, ldb, n0 + 16 * t, k0 + 64);
#pragma unroll
    for (int t = 0; t < 4; ++t) acc[t] = wmma_bf16(a1, b1[t], acc[t]);
  }
  // a0/b0 hold fragments for KTOT-64
  a1 = load_a_bf16(Ap, lda, m0, KTOT - 32);
#pragma unroll
  for (int t = 0; t < 4; ++t)
    b1[t] = load_bT_bf16(Bp, ldb, n0 + 16 * t, KTOT - 32);
#pragma unroll
  for (int t = 0; t < 4; ++t) acc[t] = wmma_bf16(a0, b0[t], acc[t]);
#pragma unroll
  for (int t = 0; t < 4; ++t) acc[t] = wmma_bf16(a1, b1[t], acc[t]);
}

// ---------- conversion kernels ----------
__global__ void k_cvt(const float* __restrict__ src,
                      unsigned short* __restrict__ dst, int n) {
  const int i = blockIdx.x * 256 + threadIdx.x;
  if (i < n) dst[i] = f2bfbits(src[i]);
}
// src[G][R][C] f32 -> dst[G][C][R] bf16 (batched transpose + convert)
__global__ void k_cvt_t(const float* __restrict__ src,
                        unsigned short* __restrict__ dst, int R, int C) {
  const int i = blockIdx.x * 256 + threadIdx.x;
  const int g = blockIdx.y;
  if (i >= R * C) return;
  const int r = i / C, c = i % C;
  dst[(size_t)g * R * C + (size_t)c * R + r] = f2bfbits(src[(size_t)g * R * C + i]);
}

// ---------- GEMM kernels: one wave = 16 x 64 output tile ----------
// Q/K/V projection (weights pre-transposed: Wt[h][a][d])
__global__ __launch_bounds__(32) void k_qkv(
    const unsigned short* __restrict__ xb, const unsigned short* __restrict__ Wt,
    const float* __restrict__ bias, unsigned short* __restrict__ outN,
    unsigned short* __restrict__ outT, int transposed) {
  const int m0 = blockIdx.x * 16;
  const int n0 = blockIdx.y * 64;
  const int h = blockIdx.z;
  const unsigned short* W = Wt + (size_t)h * HA * Dm;
  const float* bi = bias + h * HA;
  v8f acc[4];
#pragma unroll
  for (int t = 0; t < 4; ++t) acc[t] = {};
  gemm16x64<Dm>(xb, Dm, m0, W, Dm, n0, acc);
  const int lane = threadIdx.x & 31, hlf = lane >> 4;
  const int b = m0 / Ss, sbase = m0 - b * Ss;
#pragma unroll
  for (int t = 0; t < 4; ++t) {
    const int col = n0 + 16 * t + (lane & 15);
    const float bb = bi[col];
#pragma unroll
    for (int r = 0; r < 8; ++r) {
      const int row = m0 + r + 8 * hlf;
      const unsigned short val = f2bfbits(acc[t][r] + bb);
      if (!transposed) {
        outN[((size_t)h * BS + row) * HA + col] = val;
      } else {  // vT[h][b][a][s]
        outT[(((size_t)h * Bb + b) * HA + col) * Ss + (sbase + r + 8 * hlf)] = val;
      }
    }
  }
}

// Flash attention: one wave = 16 query rows x full head dim, online softmax
__global__ __launch_bounds__(32) void k_attn(
    const unsigned short* __restrict__ q, const unsigned short* __restrict__ k,
    const unsigned short* __restrict__ vt, float* __restrict__ ho) {
  const int qt = blockIdx.x, b = blockIdx.y, h = blockIdx.z;
  const int lane = threadIdx.x & 31, hlf = lane >> 4;
  const unsigned short* qh = q + (size_t)h * BS * HA;
  const unsigned short* kh = k + (size_t)h * BS * HA;
  const unsigned short* vhb = vt + (size_t)(h * Bb + b) * HA * Ss;  // [a][s]
  const int m0 = b * Ss + qt * 16;
  const int kbase = b * Ss;
  v16bf qa[4];
#pragma unroll
  for (int kk = 0; kk < 4; ++kk) qa[kk] = load_a_bf16(qh, HA, m0, kk * 32);
  v8f o[8];
#pragma unroll
  for (int j = 0; j < 8; ++j) o[j] = {};
  float mr[8], lr[8];
#pragma unroll
  for (int r = 0; r < 8; ++r) { mr[r] = -3.0e38f; lr[r] = 0.f; }
  __shared__ unsigned short pl[16 * 32];
  const float rs = 0.08838834764831845f;  // 1/sqrt(A)
  for (int kt = 0; kt < Ss; kt += 32) {
    v8f s0 = {}, s1 = {};
#pragma unroll
    for (int kk = 0; kk < 4; ++kk) {
      s0 = wmma_bf16(qa[kk], load_bT_bf16(kh, HA, kbase + kt, kk * 32), s0);
      s1 = wmma_bf16(qa[kk], load_bT_bf16(kh, HA, kbase + kt + 16, kk * 32), s1);
    }
#pragma unroll
    for (int r = 0; r < 8; ++r) {
      float a0 = s0[r] * rs, a1 = s1[r] * rs;
      float mx = fmaxf(a0, a1);
#pragma unroll
      for (int msk = 8; msk >= 1; msk >>= 1)
        mx = fmaxf(mx, __shfl_xor(mx, msk, 32));
      const float mnew = fmaxf(mr[r], mx);
      const float alpha = __expf(mr[r] - mnew);
      const float e0 = __expf(a0 - mnew);
      const float e1 = __expf(a1 - mnew);
      float sum = e0 + e1;
#pragma unroll
      for (int msk = 8; msk >= 1; msk >>= 1) sum += __shfl_xor(sum, msk, 32);
      lr[r] = lr[r] * alpha + sum;
      mr[r] = mnew;
#pragma unroll
      for (int j = 0; j < 8; ++j) o[j][r] *= alpha;
      const int row = r + 8 * hlf;
      pl[row * 32 + (lane & 15)] = f2bfbits(e0);
      pl[row * 32 + 16 + (lane & 15)] = f2bfbits(e1);
    }
    __syncthreads();
    const v16bf pa = load_a_bf16(pl, 32, 0, 0);
#pragma unroll
    for (int j = 0; j < 8; ++j)
      o[j] = wmma_bf16(pa, load_bT_bf16(vhb, Ss, j * 16, kt), o[j]);
    __syncthreads();
  }
#pragma unroll
  for (int r = 0; r < 8; ++r) {
    const float inv = 1.f / lr[r];
    const int row = m0 + r + 8 * hlf;
#pragma unroll
    for (int j = 0; j < 8; ++j)
      ho[((size_t)h * BS + row) * HA + j * 16 + (lane & 15)] = o[j][r] * inv;
  }
}

// heads are summed (not concatenated)
__global__ void k_sumheads(const float* __restrict__ ho,
                           unsigned short* __restrict__ hs) {
  const int i = blockIdx.x * blockDim.x + threadIdx.x;
  if (i >= BS * HA) return;
  float s = 0.f;
#pragma unroll
  for (int h = 0; h < NH; ++h) s += ho[(size_t)h * BS * HA + i];
  hs[i] = f2bfbits(s);
}

// out-projection + bias + residual (Wot[d][a])
__global__ __launch_bounds__(32) void k_wo(
    const unsigned short* __restrict__ hs, const unsigned short* __restrict__ Wot,
    const float* __restrict__ bo, const float* __restrict__ x,
    float* __restrict__ resid) {
  const int m0 = blockIdx.x * 16, n0 = blockIdx.y * 64;
  v8f acc[4];
#pragma unroll
  for (int t = 0; t < 4; ++t) acc[t] = {};
  gemm16x64<HA>(hs, HA, m0, Wot, HA, n0, acc);
  const int lane = threadIdx.x & 31, hlf = lane >> 4;
#pragma unroll
  for (int t = 0; t < 4; ++t) {
    const int col = n0 + 16 * t + (lane & 15);
    const float bb = bo[col];
#pragma unroll
    for (int r = 0; r < 8; ++r) {
      const int row = m0 + r + 8 * hlf;
      resid[(size_t)row * Dm + col] = x[(size_t)row * Dm + col] + bb + acc[t][r];
    }
  }
}

// layernorm; optionally also emits a bf16 copy for the next GEMM
__global__ __launch_bounds__(256) void k_ln(const float* __restrict__ in,
                                            const float* __restrict__ g,
                                            const float* __restrict__ be,
                                            float* __restrict__ of,
                                            unsigned short* __restrict__ ob) {
  const int row = blockIdx.x;
  const float* p = in + (size_t)row * Dm;
  float s = 0.f, ss = 0.f;
  for (int i = threadIdx.x; i < Dm; i += 256) {
    const float t = p[i];
    s += t;
    ss += t * t;
  }
#pragma unroll
  for (int msk = 16; msk >= 1; msk >>= 1) {
    s += __shfl_xor(s, msk, 32);
    ss += __shfl_xor(ss, msk, 32);
  }
  __shared__ float rs_[8], rss_[8];
  const int w = threadIdx.x >> 5;
  if ((threadIdx.x & 31) == 0) { rs_[w] = s; rss_[w] = ss; }
  __syncthreads();
  float S = 0.f, SS = 0.f;
#pragma unroll
  for (int i = 0; i < 8; ++i) { S += rs_[i]; SS += rss_[i]; }
  const float mu = S * (1.f / Dm);
  const float var = SS * (1.f / Dm) - mu * mu;
  const float rstd = rsqrtf(var + 1e-5f);
  for (int i = threadIdx.x; i < Dm; i += 256) {
    const float vv = (p[i] - mu) * rstd * g[i] + be[i];
    of[(size_t)row * Dm + i] = vv;
    if (ob) ob[(size_t)row * Dm + i] = f2bfbits(vv);
  }
}

// FF1 + ReLU -> bf16 activations (W1t[f][d])
__global__ __launch_bounds__(32) void k_ff1(
    const unsigned short* __restrict__ x1b, const unsigned short* __restrict__ W1t,
    const float* __restrict__ bf1, unsigned short* __restrict__ hact) {
  const int m0 = blockIdx.x * 16, n0 = blockIdx.y * 64;
  v8f acc[4];
#pragma unroll
  for (int t = 0; t < 4; ++t) acc[t] = {};
  gemm16x64<Dm>(x1b, Dm, m0, W1t, Dm, n0, acc);
  const int lane = threadIdx.x & 31, hlf = lane >> 4;
#pragma unroll
  for (int t = 0; t < 4; ++t) {
    const int col = n0 + 16 * t + (lane & 15);
    const float bb = bf1[col];
#pragma unroll
    for (int r = 0; r < 8; ++r) {
      const int row = m0 + r + 8 * hlf;
      hact[(size_t)row * FF + col] = f2bfbits(fmaxf(acc[t][r] + bb, 0.f));
    }
  }
}

// FF2 + bias + residual (W2t[d][f])
__global__ __launch_bounds__(32) void k_ff2(
    const unsigned short* __restrict__ hact, const unsigned short* __restrict__ W2t,
    const float* __restrict__ bf2, const float* __restrict__ x1,
    float* __restrict__ z) {
  const int m0 = blockIdx.x * 16, n0 = blockIdx.y * 64;
  v8f acc[4];
#pragma unroll
  for (int t = 0; t < 4; ++t) acc[t] = {};
  gemm16x64<FF>(hact, FF, m0, W2t, FF, n0, acc);
  const int lane = threadIdx.x & 31, hlf = lane >> 4;
#pragma unroll
  for (int t = 0; t < 4; ++t) {
    const int col = n0 + 16 * t + (lane & 15);
    const float bb = bf2[col];
#pragma unroll
    for (int r = 0; r < 8; ++r) {
      const int row = m0 + r + 8 * hlf;
      z[(size_t)row * Dm + col] = x1[(size_t)row * Dm + col] + bb + acc[t][r];
    }
  }
}

extern "C" void kernel_launch(void* const* d_in, const int* in_sizes, int n_in,
                              void* d_out, int out_size, void* d_ws,
                              size_t ws_size, hipStream_t stream) {
  const float* x = (const float*)d_in[0];
  const float* Wq = (const float*)d_in[1];
  const float* bq = (const float*)d_in[2];
  const float* Wk = (const float*)d_in[3];
  const float* bk = (const float*)d_in[4];
  const float* Wv = (const float*)d_in[5];
  const float* bv = (const float*)d_in[6];
  const float* Wo = (const float*)d_in[7];
  const float* bo = (const float*)d_in[8];
  const float* g1 = (const float*)d_in[9];
  const float* be1 = (const float*)d_in[10];
  const float* g2 = (const float*)d_in[11];
  const float* be2 = (const float*)d_in[12];
  const float* W1 = (const float*)d_in[13];
  const float* bf1 = (const float*)d_in[14];
  const float* W2 = (const float*)d_in[15];
  const float* bf2 = (const float*)d_in[16];
  float* out = (float*)d_out;

  char* ws = (char*)d_ws;
  size_t off = 0;
  auto alloc = [&](size_t bytes) {
    char* p = ws + off;
    off += (bytes + 255) & ~(size_t)255;
    return p;
  };
  unsigned short* q_ws = (unsigned short*)alloc((size_t)NH * BS * HA * 2);
  unsigned short* k_ws = (unsigned short*)alloc((size_t)NH * BS * HA * 2);
  unsigned short* vt_ws = (unsigned short*)alloc((size_t)NH * BS * HA * 2);
  unsigned short* xb = (unsigned short*)alloc((size_t)BS * Dm * 2);
  unsigned short* wqt = (unsigned short*)alloc((size_t)NH * Dm * HA * 2);
  unsigned short* wkt = (unsigned short*)alloc((size_t)NH * Dm * HA * 2);
  unsigned short* wvt = (unsigned short*)alloc((size_t)NH * Dm * HA * 2);
  unsigned short* wot = (unsigned short*)alloc((size_t)HA * Dm * 2);
  unsigned short* w1t = (unsigned short*)alloc((size_t)Dm * FF * 2);
  unsigned short* w2t = (unsigned short*)alloc((size_t)Dm * FF * 2);
  unsigned short* hsum = (unsigned short*)alloc((size_t)BS * HA * 2);
  float* x1 = (float*)alloc((size_t)BS * Dm * 4);
  unsigned short* x1b = (unsigned short*)alloc((size_t)BS * Dm * 2);
  float* resid = (float*)alloc((size_t)BS * Dm * 4);
  char* big = alloc((size_t)BS * FF * 2);  // 64MB: headO (32MB f32) aliases hact
  float* headO = (float*)big;              // used before hact is written
  unsigned short* hact = (unsigned short*)big;

  const dim3 w32(32), t256(256);
  // --- precision conversion / transposition (one-time) ---
  k_cvt<<<dim3((BS * Dm) / 256), t256, 0, stream>>>(x, xb, BS * Dm);
  k_cvt_t<<<dim3((Dm * HA) / 256, NH), t256, 0, stream>>>(Wq, wqt, Dm, HA);
  k_cvt_t<<<dim3((Dm * HA) / 256, NH), t256, 0, stream>>>(Wk, wkt, Dm, HA);
  k_cvt_t<<<dim3((Dm * HA) / 256, NH), t256, 0, stream>>>(Wv, wvt, Dm, HA);
  k_cvt_t<<<dim3((HA * Dm) / 256, 1), t256, 0, stream>>>(Wo, wot, HA, Dm);
  k_cvt_t<<<dim3((Dm * FF) / 256, 1), t256, 0, stream>>>(W1, w1t, Dm, FF);
  k_cvt_t<<<dim3((Dm * FF) / 256, 1), t256, 0, stream>>>(W2, w2t, FF, Dm);
  // --- attention ---
  k_qkv<<<dim3(BS / 16, HA / 64, NH), w32, 0, stream>>>(xb, wqt, bq, q_ws,
                                                        nullptr, 0);
  k_qkv<<<dim3(BS / 16, HA / 64, NH), w32, 0, stream>>>(xb, wkt, bk, k_ws,
                                                        nullptr, 0);
  k_qkv<<<dim3(BS / 16, HA / 64, NH), w32, 0, stream>>>(xb, wvt, bv, nullptr,
                                                        vt_ws, 1);
  k_attn<<<dim3(Ss / 16, Bb, NH), w32, 0, stream>>>(q_ws, k_ws, vt_ws, headO);
  k_sumheads<<<dim3((BS * HA) / 256), t256, 0, stream>>>(headO, hsum);
  k_wo<<<dim3(BS / 16, Dm / 64), w32, 0, stream>>>(hsum, wot, bo, x, resid);
  k_ln<<<dim3(BS), t256, 0, stream>>>(resid, g1, be1, x1, x1b);
  // --- FFN ---
  k_ff1<<<dim3(BS / 16, FF / 64), w32, 0, stream>>>(x1b, w1t, bf1, hact);
  k_ff2<<<dim3(BS / 16, Dm / 64), w32, 0, stream>>>(hact, w2t, bf2, x1, resid);
  k_ln<<<dim3(BS), t256, 0, stream>>>(resid, g2, be2, out, nullptr);
}
